// RNNModel_13778255086178
// MI455X (gfx1250) — matrix operand order repaired and checked by reading, not compiled, not run on hardware
//
#include <hip/hip_runtime.h>
#include <hip/hip_bf16.h>
#include <cmath>

#define B_SZ   1024
#define T_SEQ  336
#define D_IN   64
#define H_SZ   64
#define G_SZ   256
#define P_LEN  96

typedef __attribute__((ext_vector_type(16))) __bf16 bf16x16;
typedef __attribute__((ext_vector_type(8)))  __bf16 bf16x8;
typedef __attribute__((ext_vector_type(8)))  float  floatx8;

// ---------------------------------------------------------------------------
// Fast branch-free activations (native TRANS ops; no exec-mask divergence).
// ---------------------------------------------------------------------------
static __device__ __forceinline__ float fast_exp(float x) {
  // v_exp_f32 computes 2^x
  return __builtin_amdgcn_exp2f(x * 1.44269504088896340736f);
}
static __device__ __forceinline__ float fast_rcp(float x) {
  return __builtin_amdgcn_rcpf(x);
}
#if __has_builtin(__builtin_amdgcn_tanhf)
static __device__ __forceinline__ float fast_tanh(float x) {
  return __builtin_amdgcn_tanhf(x);   // native v_tanh_f32 on gfx1250
}
#else
static __device__ __forceinline__ float fast_tanh(float x) {
  // tanh(|x|) = 1 - 2*e/(1+e), e = exp(-2|x|)  (e <= 1, no overflow)
  float e = fast_exp(-2.0f * __builtin_fabsf(x));
  float r = 1.0f - 2.0f * e * fast_rcp(1.0f + e);
  return __builtin_copysignf(r, x);
}
#endif
static __device__ __forceinline__ float fast_sigmoid(float x) {
  return fast_rcp(1.0f + fast_exp(-x));
}

// ---------------------------------------------------------------------------
// WMMA fragment helpers (CDNA5 16x16x32 bf16, wave32).
//
// A (16x32 bf16) per ISA: lanes 0-15 hold row=lane, K {0..7} and {16..23};
// lanes 16-31 hold row=lane-16, K {8..15} and {24..31}.  For row-major
// activations this is exactly two contiguous b128 loads per lane.
// ---------------------------------------------------------------------------
static __device__ __forceinline__ bf16x16 frag_A_rows16(const __bf16* base, int ld, int lane) {
  const int row = lane & 15;
  const int ko  = (lane & 16) ? 8 : 0;
  const __bf16* p = base + (size_t)row * ld + ko;
  bf16x8 lo = *(const bf16x8*)(p);
  bf16x8 hi = *(const bf16x8*)(p + 16);
  bf16x16 r;
#pragma unroll
  for (int i = 0; i < 8; ++i) { r[i] = lo[i]; r[i + 8] = hi[i]; }
  return r;
}

// B (32x16 bf16) with B[k][n] = W[n][k] (weights stored [N,K] row-major, so a
// lane's 16 K-values for its column n are contiguous in memory).
// ISA: lanes 0-15 -> col n, K {0..7} and {16..23}; lanes 16-31 -> col n, K {8..15} and {24..31}.
static __device__ __forceinline__ bf16x16 frag_B_WT(const __bf16* W, int ld, int n0, int k0, int lane) {
  const int n  = n0 + (lane & 15);
  const int kb = k0 + ((lane & 16) ? 8 : 0);
  const __bf16* p = W + (size_t)n * ld + kb;
  bf16x8 lo = *(const bf16x8*)(p);
  bf16x8 hi = *(const bf16x8*)(p + 16);
  bf16x16 r;
#pragma unroll
  for (int i = 0; i < 8; ++i) { r[i] = lo[i]; r[i + 8] = hi[i]; }
  return r;
}

static __device__ __forceinline__ floatx8 wmma_bf16(bf16x16 a, bf16x16 b, floatx8 c) {
  // 8 args: (neg_a, A, neg_b, B, c_mod, C, reuse_a, reuse_b)
  return __builtin_amdgcn_wmma_f32_16x16x32_bf16(false, a, false, b, (short)0, c, false, false);
}

// ---------------------------------------------------------------------------
// Prologue: f32 -> bf16 conversion + bias pre-sum
// ---------------------------------------------------------------------------
__global__ void k_cvt_bf16(const float* __restrict__ src, __bf16* __restrict__ dst, int n) {
  int i = blockIdx.x * blockDim.x + threadIdx.x;
  if (i < n) dst[i] = (__bf16)src[i];
}

__global__ void k_bias_sum(const float* __restrict__ a, const float* __restrict__ b,
                           float* __restrict__ o, int n) {
  int i = blockIdx.x * blockDim.x + threadIdx.x;
  if (i < n) o[i] = a[i] + b[i];
}

// ---------------------------------------------------------------------------
// Persistent LSTM layer scan.  One block = 16 batch rows for all 336 steps.
// 16 waves, wave w owns gate columns [16w, 16w+16).  Weight B-fragments and
// the bias value are loop-invariant registers; x_t fragments are software-
// pipelined one step ahead so their global latency hides under the barrier
// and cell math; h_{t-1} fragments come from a 2KB LDS tile.
// ---------------------------------------------------------------------------
__global__ void __launch_bounds__(512)
k_lstm_scan(const __bf16* __restrict__ xin,   // [B, T, 64] bf16
            const __bf16* __restrict__ Wih,   // [256, 64] bf16
            const __bf16* __restrict__ Whh,   // [256, 64] bf16
            const float*  __restrict__ bias,  // [256] = b_ih + b_hh
            __bf16* __restrict__ hout)        // [B, T, 64] bf16
{
  __shared__ __bf16 h_sm[16 * H_SZ];      // h_{t-1} tile (bf16, next-step A)
  __shared__ float  pre_sm[16 * G_SZ];    // gate pre-activations (f32)

  const int tid  = threadIdx.x;
  const int lane = tid & 31;
  const int wave = tid >> 5;               // 0..15 -> gate N-tile
  const int m0   = blockIdx.x * 16;        // batch rows [m0, m0+16)
  const int n0   = wave * 16;

  // Loop-invariant weight fragments (K=0..31 and K=32..63 chunks).
  const bf16x16 bih0 = frag_B_WT(Wih, D_IN, n0, 0,  lane);
  const bf16x16 bih1 = frag_B_WT(Wih, D_IN, n0, 32, lane);
  const bf16x16 bhh0 = frag_B_WT(Whh, H_SZ, n0, 0,  lane);
  const bf16x16 bhh1 = frag_B_WT(Whh, H_SZ, n0, 32, lane);
  const float bias_n = bias[n0 + (lane & 15)];

  // Zero initial hidden state.
  for (int i = tid; i < 16 * H_SZ; i += 512) h_sm[i] = (__bf16)0.0f;

  // Each thread owns 2 of the 16x64 cell states, kept in registers.
  float c0 = 0.0f, c1 = 0.0f;
  const int e0 = tid, e1 = tid + 512;

  const size_t xrow = (size_t)m0 * T_SEQ * D_IN;   // block's first batch row
  const int    xld  = T_SEQ * D_IN;                // batch-row stride

  // Pipeline prologue: fragments for t = 0.
  bf16x16 ax0 = frag_A_rows16(xin + xrow,      xld, lane);
  bf16x16 ax1 = frag_A_rows16(xin + xrow + 32, xld, lane);

  __syncthreads();

  for (int t = 0; t < T_SEQ; ++t) {
    // ---- GEMM stage: pre = x_t @ Wih^T + h_{t-1} @ Whh^T + bias ----
    floatx8 acc;
#pragma unroll
    for (int r = 0; r < 8; ++r) acc[r] = bias_n;

    bf16x16 ah0 = frag_A_rows16(h_sm,      H_SZ, lane);
    bf16x16 ah1 = frag_A_rows16(h_sm + 32, H_SZ, lane);

    acc = wmma_bf16(ax0, bih0, acc);
    acc = wmma_bf16(ax1, bih1, acc);
    acc = wmma_bf16(ah0, bhh0, acc);
    acc = wmma_bf16(ah1, bhh1, acc);

    // Prefetch next step's x fragments (clamped; uniform control flow) —
    // issued before the barrier so the loads overlap barrier + cell math.
    {
      const int tn = (t + 1 < T_SEQ) ? (t + 1) : t;
      const __bf16* xq = xin + xrow + (size_t)tn * D_IN;
      ax0 = frag_A_rows16(xq,      xld, lane);
      ax1 = frag_A_rows16(xq + 32, xld, lane);
    }

    {
      const int n  = n0 + (lane & 15);
      const int mb = (lane & 16) ? 8 : 0;
#pragma unroll
      for (int r = 0; r < 8; ++r) pre_sm[(mb + r) * G_SZ + n] = acc[r];
    }
    __syncthreads();   // all GEMM reads of h_sm / writes of pre_sm done

    // ---- elementwise LSTM cell:  c = f*c + i*g ;  h = o*tanh(c) ----
    {
      const int row = e0 >> 6, hc = e0 & 63;
      float iv = pre_sm[row * G_SZ +       hc];
      float fv = pre_sm[row * G_SZ +  64 + hc];
      float gv = pre_sm[row * G_SZ + 128 + hc];
      float ov = pre_sm[row * G_SZ + 192 + hc];
      c0 = fast_sigmoid(fv) * c0 + fast_sigmoid(iv) * fast_tanh(gv);
      __bf16 hb = (__bf16)(fast_sigmoid(ov) * fast_tanh(c0));
      h_sm[row * H_SZ + hc] = hb;
      hout[((size_t)(m0 + row) * T_SEQ + t) * H_SZ + hc] = hb;
    }
    {
      const int row = e1 >> 6, hc = e1 & 63;
      float iv = pre_sm[row * G_SZ +       hc];
      float fv = pre_sm[row * G_SZ +  64 + hc];
      float gv = pre_sm[row * G_SZ + 128 + hc];
      float ov = pre_sm[row * G_SZ + 192 + hc];
      c1 = fast_sigmoid(fv) * c1 + fast_sigmoid(iv) * fast_tanh(gv);
      __bf16 hb = (__bf16)(fast_sigmoid(ov) * fast_tanh(c1));
      h_sm[row * H_SZ + hc] = hb;
      hout[((size_t)(m0 + row) * T_SEQ + t) * H_SZ + hc] = hb;
    }
    __syncthreads();   // new h visible before next GEMM stage
  }
}

// ---------------------------------------------------------------------------
// Autoregressive decoder: 96 steps of (zero-state LSTM step x2) + FC,
// feeding the prediction back through LDS as bf16.
// ---------------------------------------------------------------------------
__global__ void __launch_bounds__(512)
k_decoder(const __bf16* __restrict__ h2,     // [B, T, 64] encoder layer-1 h (bf16)
          const __bf16* __restrict__ Wih0, const float* __restrict__ bias0,
          const __bf16* __restrict__ Wih1, const float* __restrict__ bias1,
          const __bf16* __restrict__ Wfc,  const float* __restrict__ bfc,
          float* __restrict__ out)           // [B, 96, 64] f32
{
  __shared__ __bf16 cur_sm[16 * H_SZ];    // current input (pred feedback)
  __shared__ __bf16 hid_sm[16 * H_SZ];    // intermediate hidden
  __shared__ float  pre_sm[16 * G_SZ];

  const int tid  = threadIdx.x;
  const int lane = tid & 31;
  const int wave = tid >> 5;
  const int m0   = blockIdx.x * 16;
  const int n0   = wave * 16;
  const int e0 = tid, e1 = tid + 512;

  const bf16x16 b00 = frag_B_WT(Wih0, D_IN, n0, 0,  lane);
  const bf16x16 b01 = frag_B_WT(Wih0, D_IN, n0, 32, lane);
  const bf16x16 b10 = frag_B_WT(Wih1, H_SZ, n0, 0,  lane);
  const bf16x16 b11 = frag_B_WT(Wih1, H_SZ, n0, 32, lane);
  const float bias0_n = bias0[n0 + (lane & 15)];
  const float bias1_n = bias1[n0 + (lane & 15)];

  bf16x16 bf0 = {}, bf1 = {};
  float bfc_n = 0.0f;
  if (wave < 4) {                          // FC output has only 64 columns
    bf0 = frag_B_WT(Wfc, H_SZ, n0, 0,  lane);
    bf1 = frag_B_WT(Wfc, H_SZ, n0, 32, lane);
    bfc_n = bfc[n0 + (lane & 15)];
  }

  // current = h2[:, T-1, :]
  for (int i = tid; i < 16 * H_SZ; i += 512) {
    int row = i >> 6, hc = i & 63;
    cur_sm[i] = h2[((size_t)(m0 + row) * T_SEQ + (T_SEQ - 1)) * H_SZ + hc];
  }
  __syncthreads();

  // pre = A @ W^T + bias  ->  pre_sm
  auto gate_stage = [&](const __bf16* Abuf, const bf16x16& w0, const bf16x16& w1, float bn) {
    floatx8 acc;
#pragma unroll
    for (int r = 0; r < 8; ++r) acc[r] = bn;
    bf16x16 a0 = frag_A_rows16(Abuf,      H_SZ, lane);
    bf16x16 a1 = frag_A_rows16(Abuf + 32, H_SZ, lane);
    acc = wmma_bf16(a0, w0, acc);
    acc = wmma_bf16(a1, w1, acc);
    const int n  = n0 + (lane & 15);
    const int mb = (lane & 16) ? 8 : 0;
#pragma unroll
    for (int r = 0; r < 8; ++r) pre_sm[(mb + r) * G_SZ + n] = acc[r];
  };

  // zero-state cell:  c = i*g ;  h = o*tanh(c)   (forget gate unused)
  auto cell_zero = [&](int e, __bf16* dst) {
    const int row = e >> 6, hc = e & 63;
    float iv = pre_sm[row * G_SZ +       hc];
    float gv = pre_sm[row * G_SZ + 128 + hc];
    float ov = pre_sm[row * G_SZ + 192 + hc];
    float c  = fast_sigmoid(iv) * fast_tanh(gv);
    dst[row * H_SZ + hc] = (__bf16)(fast_sigmoid(ov) * fast_tanh(c));
  };

  for (int t = 0; t < P_LEN; ++t) {
    gate_stage(cur_sm, b00, b01, bias0_n);          // layer 0
    __syncthreads();
    cell_zero(e0, hid_sm); cell_zero(e1, hid_sm);
    __syncthreads();

    gate_stage(hid_sm, b10, b11, bias1_n);          // layer 1
    __syncthreads();
    cell_zero(e0, hid_sm); cell_zero(e1, hid_sm);   // overwrite with h2 (safe post-barrier)
    __syncthreads();

    if (wave < 4) {                                 // FC: pred = h2 @ Wfc^T + bfc
      floatx8 acc;
#pragma unroll
      for (int r = 0; r < 8; ++r) acc[r] = bfc_n;
      bf16x16 a0 = frag_A_rows16(hid_sm,      H_SZ, lane);
      bf16x16 a1 = frag_A_rows16(hid_sm + 32, H_SZ, lane);
      acc = wmma_bf16(a0, bf0, acc);
      acc = wmma_bf16(a1, bf1, acc);
      const int n  = n0 + (lane & 15);
      const int mb = (lane & 16) ? 8 : 0;
#pragma unroll
      for (int r = 0; r < 8; ++r) {
        const int m = mb + r;
        const float v = acc[r];
        out[(size_t)(m0 + m) * P_LEN * D_IN + (size_t)t * D_IN + n] = v;
        cur_sm[m * D_IN + n] = (__bf16)v;           // feedback for next step
      }
    }
    __syncthreads();
  }
}

// ---------------------------------------------------------------------------
extern "C" void kernel_launch(void* const* d_in, const int* in_sizes, int n_in,
                              void* d_out, int out_size, void* d_ws, size_t ws_size,
                              hipStream_t stream) {
  (void)in_sizes; (void)n_in; (void)out_size; (void)ws_size;

  const float* x     = (const float*)d_in[0];
  const float* W_ih0 = (const float*)d_in[1];
  const float* W_hh0 = (const float*)d_in[2];
  const float* b_ih0 = (const float*)d_in[3];
  const float* b_hh0 = (const float*)d_in[4];
  const float* W_ih1 = (const float*)d_in[5];
  const float* W_hh1 = (const float*)d_in[6];
  const float* b_ih1 = (const float*)d_in[7];
  const float* b_hh1 = (const float*)d_in[8];
  const float* W_fc  = (const float*)d_in[9];
  const float* b_fc  = (const float*)d_in[10];
  float* out = (float*)d_out;

  char* ws = (char*)d_ws;
  size_t off = 0;
  auto carve = [&](size_t bytes) {
    char* p = ws + off;
    off += (bytes + 255) & ~(size_t)255;
    return p;
  };
  __bf16* x_bf  = (__bf16*)carve((size_t)B_SZ * T_SEQ * D_IN * 2);
  __bf16* h1_bf = (__bf16*)carve((size_t)B_SZ * T_SEQ * H_SZ * 2);
  __bf16* h2_bf = (__bf16*)carve((size_t)B_SZ * T_SEQ * H_SZ * 2);
  __bf16* Wih0b = (__bf16*)carve((size_t)G_SZ * D_IN * 2);
  __bf16* Whh0b = (__bf16*)carve((size_t)G_SZ * H_SZ * 2);
  __bf16* Wih1b = (__bf16*)carve((size_t)G_SZ * H_SZ * 2);
  __bf16* Whh1b = (__bf16*)carve((size_t)G_SZ * H_SZ * 2);
  __bf16* Wfcb  = (__bf16*)carve((size_t)D_IN * H_SZ * 2);
  float*  bias0 = (float*)carve(G_SZ * sizeof(float));
  float*  bias1 = (float*)carve(G_SZ * sizeof(float));

  const int nx = B_SZ * T_SEQ * D_IN;
  k_cvt_bf16<<<(nx + 255) / 256, 256, 0, stream>>>(x, x_bf, nx);
  k_cvt_bf16<<<(G_SZ * D_IN + 255) / 256, 256, 0, stream>>>(W_ih0, Wih0b, G_SZ * D_IN);
  k_cvt_bf16<<<(G_SZ * H_SZ + 255) / 256, 256, 0, stream>>>(W_hh0, Whh0b, G_SZ * H_SZ);
  k_cvt_bf16<<<(G_SZ * H_SZ + 255) / 256, 256, 0, stream>>>(W_ih1, Wih1b, G_SZ * H_SZ);
  k_cvt_bf16<<<(G_SZ * H_SZ + 255) / 256, 256, 0, stream>>>(W_hh1, Whh1b, G_SZ * H_SZ);
  k_cvt_bf16<<<(D_IN * H_SZ + 255) / 256, 256, 0, stream>>>(W_fc, Wfcb, D_IN * H_SZ);
  k_bias_sum<<<1, 256, 0, stream>>>(b_ih0, b_hh0, bias0, G_SZ);
  k_bias_sum<<<1, 256, 0, stream>>>(b_ih1, b_hh1, bias1, G_SZ);

  // Encoder: two persistent LSTM layers over the full sequence.
  k_lstm_scan<<<B_SZ / 16, 512, 0, stream>>>(x_bf,  Wih0b, Whh0b, bias0, h1_bf);
  k_lstm_scan<<<B_SZ / 16, 512, 0, stream>>>(h1_bf, Wih1b, Whh1b, bias1, h2_bf);

  // Decoder: 96 autoregressive steps, batch-parallel.
  k_decoder<<<B_SZ / 16, 512, 0, stream>>>(h2_bf, Wih0b, bias0, Wih1b, bias1,
                                           Wfcb, b_fc, out);
}
